// Block_51642686767350
// MI455X (gfx1250) — compile-verified, matrix operand order checked by reading, not run on hardware
//
#include <hip/hip_runtime.h>
#include <hip/hip_bf16.h>
#include <math.h>

#define Bq 8
#define Nq 128
#define Cq 256
#define Hq 8
#define HSq 32
#define SCALEq 0.17677669529663687f
#define EPSq 1e-5f

typedef unsigned short ushort_t;
typedef __attribute__((ext_vector_type(16))) __bf16 v16bf;
typedef __attribute__((ext_vector_type(8)))  float  v8f;

union FragU { v16bf v; uint4 u[2]; };
union FragCvt { v16bf v; unsigned int w[8]; };

__device__ __forceinline__ ushort_t f2bf(float x) {
  union { float f; unsigned int u; } cv; cv.f = x;
  unsigned int r = cv.u + 0x7FFFu + ((cv.u >> 16) & 1u);
  return (ushort_t)(r >> 16);
}

// Load a 16x32 bf16 WMMA operand fragment from a row-major [rows][ld] bf16
// matrix, tile origin (r0, k0). Matches CDNA5 16-bit A-matrix layout:
// lane<16: row=lane,    K = k0+{0..7, 16..23}
// lane>=16: row=lane-16, K = k0+{8..15, 24..31}
__device__ __forceinline__ v16bf load_frag(const ushort_t* base, int ld, int r0, int k0, int lane) {
  const int r   = r0 + (lane & 15);
  const int sel = (lane >> 4) & 1;
  const ushort_t* p = base + (size_t)r * ld + (k0 + sel * 8);
  FragU f;
  f.u[0] = *(const uint4*)(p);
  f.u[1] = *(const uint4*)(p + 16);
  return f.v;
}

__device__ __forceinline__ v8f wmma_bf16(v16bf a, v16bf b, v8f c) {
  return __builtin_amdgcn_wmma_f32_16x16x32_bf16(false, a, false, b, (short)0, c, false, false);
}

__device__ __forceinline__ float wred_sum(float v) {
#pragma unroll
  for (int m = 16; m >= 1; m >>= 1) v += __shfl_xor(v, m, 32);
  return v;
}
__device__ __forceinline__ float wred_max(float v) {
#pragma unroll
  for (int m = 16; m >= 1; m >>= 1) v = fmaxf(v, __shfl_xor(v, m, 32));
  return v;
}

// ---------------------------------------------------------------- prep ----
__global__ void k_f32_to_bf16(const float* __restrict__ src, ushort_t* __restrict__ dst, int n) {
  for (int i = blockIdx.x * blockDim.x + threadIdx.x; i < n; i += gridDim.x * blockDim.x)
    dst[i] = f2bf(src[i]);
}

// Wsum[h] = sum_c ln2_w[c]*rconv[h,c];  bsum[h] = sum_c ln2_b[c]*rconv[h,c]
__global__ void k_prep_sums(const float* __restrict__ ln2w, const float* __restrict__ ln2b,
                            const float* __restrict__ rconv,
                            float* __restrict__ Wsum, float* __restrict__ bsum) {
  const int lane = threadIdx.x & 31;
  const int h    = threadIdx.x >> 5;  // 8 waves
  float sW = 0.f, sB = 0.f;
  for (int c = lane; c < Cq; c += 32) {
    float r = rconv[h * Cq + c];
    sW += ln2w[c] * r;
    sB += ln2b[c] * r;
  }
  sW = wred_sum(sW);
  sB = wred_sum(sB);
  if (lane == 0) { Wsum[h] = sW; bsum[h] = sB; }
}

// w2b: row-major [16][256] bf16. Rows 0..7 = ln2_w[c]*rconv[h,c] (the folded
// LN-scale x rconv weights), row 8 = 1.0 (ones column -> sum(x)), rows 9..15 = 0.
__global__ void k_prep_w2b(const float* __restrict__ ln2w, const float* __restrict__ rconv,
                           ushort_t* __restrict__ w2b) {
  const int c = threadIdx.x;  // 256 threads
#pragma unroll
  for (int n = 0; n < 16; ++n) {
    float v = (n < 8) ? ln2w[c] * rconv[n * Cq + c] : (n == 8 ? 1.f : 0.f);
    w2b[n * Cq + c] = f2bf(v);
  }
}

// ----------------------------------------------- LayerNorm rows -> bf16 ---
__global__ void k_ln_rows_bf16(const float* __restrict__ x, const float* __restrict__ gw,
                               const float* __restrict__ gb, ushort_t* __restrict__ out, int rows) {
  const int lane = threadIdx.x & 31;
  const int w    = threadIdx.x >> 5;
  const int row  = blockIdx.x * 8 + w;
  if (row >= rows) return;
  const float* p = x + (size_t)row * Cq + lane * 8;
  float4 a = *(const float4*)p;
  float4 b = *(const float4*)(p + 4);
  float xv[8] = {a.x, a.y, a.z, a.w, b.x, b.y, b.z, b.w};
  float S = 0.f, Q = 0.f;
#pragma unroll
  for (int i = 0; i < 8; ++i) { S += xv[i]; Q += xv[i] * xv[i]; }
  S = wred_sum(S); Q = wred_sum(Q);
  float mu   = S * (1.f / Cq);
  float rstd = rsqrtf(Q * (1.f / Cq) - mu * mu + EPSq);
  float4 w0 = *(const float4*)(gw + lane * 8);
  float4 w1 = *(const float4*)(gw + lane * 8 + 4);
  float4 b0 = *(const float4*)(gb + lane * 8);
  float4 b1 = *(const float4*)(gb + lane * 8 + 4);
  float gwv[8] = {w0.x, w0.y, w0.z, w0.w, w1.x, w1.y, w1.z, w1.w};
  float gbv[8] = {b0.x, b0.y, b0.z, b0.w, b1.x, b1.y, b1.z, b1.w};
  union { ushort_t h[8]; uint4 u; } ob;
#pragma unroll
  for (int i = 0; i < 8; ++i) ob.h[i] = f2bf((xv[i] - mu) * rstd * gwv[i] + gbv[i]);
  *(uint4*)(out + (size_t)row * Cq + lane * 8) = ob.u;
}

// ------------- dominant streaming pass: LN2 + rconv -> attn_R[B,H,N,N] ----
// WMMA formulation: per wave, one tile of 16 relation rows x 256 channels.
// A = rows (f32 -> bf16 packed via v_perm), B = constant w2b (cols 0..7 =
// per-head dots D[h], col 8 = sum(x)). Only sum(x^2) stays in VALU, split
// across 4 accumulators to avoid a serial FMA dependence chain.
__global__ void k_relation(const float* __restrict__ rel, const ushort_t* __restrict__ w2b,
                           const float* __restrict__ Wsum, const float* __restrict__ bsum,
                           float* __restrict__ attnR) {
  const int lane = threadIdx.x & 31;
  const int w    = threadIdx.x >> 5;  // 8 waves / block
  const int col  = lane & 15;         // fragment column (head h for col<8)
  const int hi   = lane >> 4;

  // hoist the 8 loop-invariant B fragments (w2b is [16][256] bf16)
  v16bf bfr[8];
#pragma unroll
  for (int ks = 0; ks < 8; ++ks) bfr[ks] = load_frag(w2b, Cq, 0, ks * 32, lane);

  const float Wl = Wsum[col & 7];
  const float Bl = bsum[col & 7];

  const int totalTiles = (Bq * Nq * Nq) / 16;  // 8192 tiles of 16 rows
  const int stride     = gridDim.x * 8;
  for (int tile = blockIdx.x * 8 + w; tile < totalTiles; tile += stride) {
    const int r = tile * 16 + col;  // this lane's A-matrix row (global row id)
    const float* prow = rel + (size_t)r * Cq + hi * 8;
    __builtin_prefetch(rel + ((size_t)r + (size_t)stride * 16) * Cq + hi * 8, 0, 0);
    float Qa[4] = {0.f, 0.f, 0.f, 0.f};
    v8f c = {};
#pragma unroll
    for (int ks = 0; ks < 8; ++ks) {
      const float* p = prow + ks * 32;
      float4 x0 = *(const float4*)(p);
      float4 x1 = *(const float4*)(p + 4);
      float4 x2 = *(const float4*)(p + 16);
      float4 x3 = *(const float4*)(p + 20);
      float xv[16] = {x0.x, x0.y, x0.z, x0.w, x1.x, x1.y, x1.z, x1.w,
                      x2.x, x2.y, x2.z, x2.w, x3.x, x3.y, x3.z, x3.w};
      FragCvt fa;
#pragma unroll
      for (int i = 0; i < 8; ++i) {  // pack pairs of f32 -> bf16x2, 1 v_perm each
        union { float f; unsigned int u; } lo, hic;
        lo.f = xv[2 * i]; hic.f = xv[2 * i + 1];
        fa.w[i] = __builtin_amdgcn_perm(hic.u, lo.u, 0x07060302u);
      }
#pragma unroll
      for (int i = 0; i < 16; ++i) Qa[i & 3] = fmaf(xv[i], xv[i], Qa[i & 3]);
      c = wmma_bf16(fa.v, bfr[ks], c);
    }
    float Q = (Qa[0] + Qa[1]) + (Qa[2] + Qa[3]);
    Q += __shfl_xor(Q, 16, 32);  // full sum(x^2) for row (lane&15)

#pragma unroll
    for (int j = 0; j < 8; ++j) {
      const int rloc = j + (hi << 3);                  // local row of c[j]
      float Sj = __shfl(c[j], 8 | (lane & 16), 32);    // ones-column -> sum(x)
      float Qj = __shfl(Q, rloc, 32);                  // lane rloc owns row rloc
      float mu   = Sj * (1.f / Cq);
      float rstd = rsqrtf(Qj * (1.f / Cq) - mu * mu + EPSq);
      float val  = rstd * (c[j] - mu * Wl) + Bl;
      if (col < 8) {
        int g = tile * 16 + rloc;
        int b = g >> 14, n = (g >> 7) & 127, m = g & 127;
        attnR[(((size_t)(b * Hq + col) * Nq + n) * Nq) + m] = val;
      }
    }
  }
}

// ------------------------------------------------ generic WMMA GEMM -------
// C[M,N] = A[M,K](bf16) @ Bw[N,K](bf16)^T, fused epilogues.
#define GEMM_STORE_BF16 0
#define GEMM_BIAS_RES_F32 1
#define GEMM_BIAS_GELU_BF16 2
__global__ void k_gemm_wmma(const ushort_t* __restrict__ A, const ushort_t* __restrict__ Bw,
                            int K, int N, const float* __restrict__ bias,
                            const float* __restrict__ resid, float* __restrict__ outF,
                            ushort_t* __restrict__ outB, int mode) {
  const int lane = threadIdx.x & 31;
  const int w    = threadIdx.x >> 5;
  const int m0   = blockIdx.x * 16;
  const int n0   = (blockIdx.y * 8 + w) * 16;
  v8f c = {};
  for (int k0 = 0; k0 < K; k0 += 32) {
    v16bf a = load_frag(A, K, m0, k0, lane);
    v16bf b = load_frag(Bw, K, n0, k0, lane);
    c = wmma_bf16(a, b, c);
  }
  const int col   = n0 + (lane & 15);
  const int rbase = m0 + ((lane >> 4) << 3);
  float bval = (mode != GEMM_STORE_BF16) ? bias[col] : 0.f;
#pragma unroll
  for (int j = 0; j < 8; ++j) {
    int row = rbase + j;
    float v = c[j];
    size_t idx = (size_t)row * N + col;
    if (mode == GEMM_STORE_BF16) {
      outB[idx] = f2bf(v);
    } else if (mode == GEMM_BIAS_RES_F32) {
      outF[idx] = resid[idx] + v + bval;
    } else {  // bias + exact GELU -> bf16
      float t = v + bval;
      float ge = 0.5f * t * (1.f + erff(t * 0.70710678118654752f));
      outB[idx] = f2bf(ge);
    }
  }
}

// --------------------------- fused attention core, one block per (b,h) ----
__global__ void k_attn(const ushort_t* __restrict__ qkvb, const float* __restrict__ attnR,
                       const float* __restrict__ conn, ushort_t* __restrict__ xb) {
  const int bh = blockIdx.x;
  const int b  = bh >> 3;
  const int h  = bh & 7;
  const int tid  = threadIdx.x;
  const int lane = tid & 31;
  const int w    = tid >> 5;  // 8 waves, wave owns rows n0..n0+15
  const int n0   = w * 16;

  __shared__ alignas(16) unsigned char smem[65536];
  float*    attnS = (float*)smem;                 // [128][128] f32 scores
  ushort_t* probs = (ushort_t*)smem;              // [128][128] bf16 (aliases first 32KB)
  ushort_t* vt    = (ushort_t*)(smem + 32768);    // [32][128]  bf16 v^T

  const ushort_t* qbase = qkvb + (size_t)b * Nq * 768 + h * HSq;
  const ushort_t* kbase = qbase + 256;
  const ushort_t* vbase = qbase + 512;

  // --- attn_J = q @ k^T : K=32 -> one WMMA per 16x16 tile ---
  v16bf aq = load_frag(qbase, 768, n0, 0, lane);
#pragma unroll
  for (int t = 0; t < 8; ++t) {
    v16bf bk = load_frag(kbase, 768, t * 16, 0, lane);
    v8f c = {};
    c = wmma_bf16(aq, bk, c);
#pragma unroll
    for (int j = 0; j < 8; ++j)
      attnS[(n0 + j + ((lane >> 4) << 3)) * Nq + t * 16 + (lane & 15)] = c[j];
  }
  __syncthreads();

  // --- softmax over m, fused with (attn_J + attn_R) * conn * SCALE ---
  const float* aRbase = attnR + (size_t)(b * Hq + h) * Nq * Nq;
  const float* cnbase = conn + (size_t)b * Nq * Nq;
  float pbuf[64];
#pragma unroll
  for (int rr = 0; rr < 16; ++rr) {
    const int r = n0 + rr;
    float4 aJ = *(const float4*)(attnS + r * Nq + lane * 4);
    float4 aR = *(const float4*)(aRbase + r * Nq + lane * 4);
    float4 cn = *(const float4*)(cnbase + r * Nq + lane * 4);
    float z[4];
    z[0] = (aJ.x + aR.x) * cn.x * SCALEq;
    z[1] = (aJ.y + aR.y) * cn.y * SCALEq;
    z[2] = (aJ.z + aR.z) * cn.z * SCALEq;
    z[3] = (aJ.w + aR.w) * cn.w * SCALEq;
    float mx = fmaxf(fmaxf(z[0], z[1]), fmaxf(z[2], z[3]));
    mx = wred_max(mx);
    float e[4], s = 0.f;
#pragma unroll
    for (int i = 0; i < 4; ++i) { e[i] = __expf(z[i] - mx); s += e[i]; }
    s = wred_sum(s);
    float inv = 1.f / s;
#pragma unroll
    for (int i = 0; i < 4; ++i) pbuf[rr * 4 + i] = e[i] * inv;
  }
  __syncthreads();  // all reads of f32 scores done before bf16 overwrite

  // write probabilities as bf16 (lower 32KB), stage v^T into LDS (next 8KB)
#pragma unroll
  for (int rr = 0; rr < 16; ++rr) {
    const int r = n0 + rr;
#pragma unroll
    for (int i = 0; i < 4; ++i) probs[r * Nq + lane * 4 + i] = f2bf(pbuf[rr * 4 + i]);
  }
  for (int i = tid; i < Nq * HSq; i += 256) {
    int m = i >> 5, d = i & 31;
    vt[d * Nq + m] = vbase[(size_t)m * 768 + d];
  }
  __syncthreads();

  // --- out = probs @ v : [128,128] x [128,32], K=128 -> 4 WMMA per tile ---
#pragma unroll
  for (int dt = 0; dt < 2; ++dt) {
    v8f c = {};
#pragma unroll
    for (int k0 = 0; k0 < Nq; k0 += 32) {
      v16bf ap = load_frag(probs, Nq, n0, k0, lane);
      v16bf bv = load_frag(vt, Nq, dt * 16, k0, lane);
      c = wmma_bf16(ap, bv, c);
    }
    const int d = dt * 16 + (lane & 15);
#pragma unroll
    for (int j = 0; j < 8; ++j) {
      const int n = n0 + j + ((lane >> 4) << 3);
      xb[((size_t)(b * Nq + n)) * Cq + h * HSq + d] = f2bf(c[j]);
    }
  }
}

// --------------------------------------------------------------- launch ---
extern "C" void kernel_launch(void* const* d_in, const int* in_sizes, int n_in,
                              void* d_out, int out_size, void* d_ws, size_t ws_size,
                              hipStream_t stream) {
  const float* joint = (const float*)d_in[0];
  const float* rel   = (const float*)d_in[1];
  const float* conn  = (const float*)d_in[2];
  const float* ln1w  = (const float*)d_in[3];
  const float* ln1b  = (const float*)d_in[4];
  const float* ln2w  = (const float*)d_in[5];
  const float* ln2b  = (const float*)d_in[6];
  const float* ln3w  = (const float*)d_in[7];
  const float* ln3b  = (const float*)d_in[8];
  const float* qkv_w = (const float*)d_in[9];
  const float* rconv = (const float*)d_in[10];
  const float* projw = (const float*)d_in[11];
  const float* projb = (const float*)d_in[12];
  const float* fc1w  = (const float*)d_in[13];
  const float* fc1b  = (const float*)d_in[14];
  const float* fc2w  = (const float*)d_in[15];
  const float* fc2b  = (const float*)d_in[16];

  char* p = (char*)d_ws;
  ushort_t* qkv_wb = (ushort_t*)p;            p += (size_t)768 * 256 * 2;
  ushort_t* projwb = (ushort_t*)p;            p += (size_t)256 * 256 * 2;
  ushort_t* fc1wb  = (ushort_t*)p;            p += (size_t)256 * 256 * 2;
  ushort_t* fc2wb  = (ushort_t*)p;            p += (size_t)256 * 256 * 2;
  float*    Wsum   = (float*)p;               p += 256;
  float*    bsum   = (float*)p;               p += 256;
  ushort_t* w2b    = (ushort_t*)p;            p += (size_t)16 * 256 * 2;
  ushort_t* jn_bf  = (ushort_t*)p;            p += (size_t)1024 * 256 * 2;
  ushort_t* qkv_bf = (ushort_t*)p;            p += (size_t)1024 * 768 * 2;
  float*    attnR  = (float*)p;               p += (size_t)8 * 8 * 128 * 128 * 4;
  ushort_t* x_bf   = (ushort_t*)p;            p += (size_t)1024 * 256 * 2;
  float*    jointW = (float*)p;               p += (size_t)1024 * 256 * 4;
  ushort_t* jn3_bf = (ushort_t*)p;            p += (size_t)1024 * 256 * 2;
  ushort_t* h1_bf  = (ushort_t*)p;            p += (size_t)1024 * 256 * 2;

  // weight conversions + folded LN2 products/sums
  k_f32_to_bf16<<<192, 256, 0, stream>>>(qkv_w, qkv_wb, 768 * 256);
  k_f32_to_bf16<<<64, 256, 0, stream>>>(projw, projwb, 256 * 256);
  k_f32_to_bf16<<<64, 256, 0, stream>>>(fc1w, fc1wb, 256 * 256);
  k_f32_to_bf16<<<64, 256, 0, stream>>>(fc2w, fc2wb, 256 * 256);
  k_prep_sums<<<1, 256, 0, stream>>>(ln2w, ln2b, rconv, Wsum, bsum);
  k_prep_w2b<<<1, 256, 0, stream>>>(ln2w, rconv, w2b);

  // attention branch
  k_ln_rows_bf16<<<128, 256, 0, stream>>>(joint, ln1w, ln1b, jn_bf, 1024);
  k_relation<<<512, 256, 0, stream>>>(rel, w2b, Wsum, bsum, attnR);
  k_gemm_wmma<<<dim3(64, 6), 256, 0, stream>>>(jn_bf, qkv_wb, 256, 768, nullptr, nullptr,
                                               nullptr, qkv_bf, GEMM_STORE_BF16);
  k_attn<<<64, 256, 0, stream>>>(qkv_bf, attnR, conn, x_bf);
  k_gemm_wmma<<<dim3(64, 2), 256, 0, stream>>>(x_bf, projwb, 256, 256, projb, joint,
                                               jointW, nullptr, GEMM_BIAS_RES_F32);

  // MLP branch
  k_ln_rows_bf16<<<128, 256, 0, stream>>>(jointW, ln3w, ln3b, jn3_bf, 1024);
  k_gemm_wmma<<<dim3(64, 2), 256, 0, stream>>>(jn3_bf, fc1wb, 256, 256, fc1b, nullptr,
                                               nullptr, h1_bf, GEMM_BIAS_GELU_BF16);
  k_gemm_wmma<<<dim3(64, 2), 256, 0, stream>>>(h1_bf, fc2wb, 256, 256, fc2b, jointW,
                                               (float*)d_out, nullptr, GEMM_BIAS_RES_F32);
}
